// CausalSelfAttentionSingleHead_17626545782951
// MI455X (gfx1250) — compile-verified, matrix operand order checked by reading, not run on hardware
//
#include <hip/hip_runtime.h>
#include <hip/hip_bf16.h>

typedef __attribute__((ext_vector_type(16))) __bf16 bf16x16;
typedef __attribute__((ext_vector_type(8)))  float  f32x8;

#define D_MODEL 1024
#define D_HEAD  64
#define BATCH   4
#define SEQ     4096
#define NROWS   (BATCH * SEQ)
#define KBLK    64

union B16Frag { uint4 u[2]; bf16x16 v; };

// A/B fragment loader from bf16 row-major source (rows = lanes' M or N index).
// Layout per ISA 7.12.2 (16-bit 16x32): lanes 0-15 -> K 0..7 & 16..23,
// lanes 16-31 -> K 8..15 & 24..31.
__device__ __forceinline__ bf16x16 load_frag_b16(const __bf16* base, int rstride, int lane) {
    const int h = (lane >> 4) & 1;
    const int r = lane & 15;
    const __bf16* p = base + (size_t)r * rstride + 8 * h;
    B16Frag t;
    t.u[0] = *reinterpret_cast<const uint4*>(p);
    t.u[1] = *reinterpret_cast<const uint4*>(p + 16);
    return t.v;
}

// Same fragment pattern but reading fp32 and converting to bf16 in registers.
__device__ __forceinline__ bf16x16 load_frag_f32(const float* base, int rstride, int lane) {
    const int h = (lane >> 4) & 1;
    const int r = lane & 15;
    const float* p = base + (size_t)r * rstride + 8 * h;
    float4 a = *reinterpret_cast<const float4*>(p);
    float4 b = *reinterpret_cast<const float4*>(p + 4);
    float4 c = *reinterpret_cast<const float4*>(p + 16);
    float4 d = *reinterpret_cast<const float4*>(p + 20);
    bf16x16 v;
    v[0]  = (__bf16)a.x; v[1]  = (__bf16)a.y; v[2]  = (__bf16)a.z; v[3]  = (__bf16)a.w;
    v[4]  = (__bf16)b.x; v[5]  = (__bf16)b.y; v[6]  = (__bf16)b.z; v[7]  = (__bf16)b.w;
    v[8]  = (__bf16)c.x; v[9]  = (__bf16)c.y; v[10] = (__bf16)c.z; v[11] = (__bf16)c.w;
    v[12] = (__bf16)d.x; v[13] = (__bf16)d.y; v[14] = (__bf16)d.z; v[15] = (__bf16)d.w;
    return v;
}

__device__ __forceinline__ f32x8 wmma_bf16(bf16x16 a, bf16x16 b, f32x8 c) {
    return __builtin_amdgcn_wmma_f32_16x16x32_bf16(false, a, false, b, (short)0, c, false, false);
}

__device__ __forceinline__ float rowmax16(float v) {
    v = fmaxf(v, __shfl_xor(v, 1, 16));
    v = fmaxf(v, __shfl_xor(v, 2, 16));
    v = fmaxf(v, __shfl_xor(v, 4, 16));
    v = fmaxf(v, __shfl_xor(v, 8, 16));
    return v;
}
__device__ __forceinline__ float rowsum16(float v) {
    v += __shfl_xor(v, 1, 16);
    v += __shfl_xor(v, 2, 16);
    v += __shfl_xor(v, 4, 16);
    v += __shfl_xor(v, 8, 16);
    return v;
}

// Async global->LDS 16-byte copy (CDNA5 path, tracked by ASYNCcnt).
__device__ __forceinline__ void async_copy16(const __bf16* g, __bf16* l) {
    unsigned loff = (unsigned)(unsigned long long)(const void*)l;  // low 32b = LDS offset
    asm volatile("global_load_async_to_lds_b128 %0, %1, off"
                 :: "v"(loff), "v"((unsigned long long)g) : "memory");
}

// ---------------- Kernel 1: W -> W^T (bf16) -------------------------------
__global__ void wt_prep_kernel(const float* __restrict__ Wq,
                               const float* __restrict__ Wk,
                               const float* __restrict__ Wv,
                               __bf16* __restrict__ Wt) {
    int idx = blockIdx.x * blockDim.x + threadIdx.x;
    if (idx >= 3 * D_HEAD * D_MODEL) return;
    int m    = idx / (D_HEAD * D_MODEL);
    int rem  = idx % (D_HEAD * D_MODEL);
    int hcol = rem / D_MODEL;
    int k    = rem % D_MODEL;
    const float* W = (m == 0) ? Wq : ((m == 1) ? Wk : Wv);
    Wt[idx] = (__bf16)W[(size_t)k * D_HEAD + hcol];
}

// ---------------- Kernel 2: fused QKV projection (WMMA bf16) --------------
__global__ void __launch_bounds__(128)
qkv_proj_kernel(const float* __restrict__ x, const __bf16* __restrict__ Wt,
                const float* __restrict__ bq, const float* __restrict__ bk,
                const float* __restrict__ bv,
                __bf16* __restrict__ Qb, __bf16* __restrict__ Kb,
                __bf16* __restrict__ Vt) {
    const int lane = threadIdx.x & 31;
    const int wave = threadIdx.x >> 5;       // n-tile 0..3
    const int row0 = blockIdx.x * 16;
    const int n0   = wave * 16;

    f32x8 accq = {}, acck = {}, accv = {};
    const __bf16* wtq = Wt + (size_t)0 * D_HEAD * D_MODEL + (size_t)n0 * D_MODEL;
    const __bf16* wtk = Wt + (size_t)1 * D_HEAD * D_MODEL + (size_t)n0 * D_MODEL;
    const __bf16* wtv = Wt + (size_t)2 * D_HEAD * D_MODEL + (size_t)n0 * D_MODEL;

    for (int k0 = 0; k0 < D_MODEL; k0 += 32) {
        bf16x16 a  = load_frag_f32(x + (size_t)row0 * D_MODEL + k0, D_MODEL, lane);
        bf16x16 b0 = load_frag_b16(wtq + k0, D_MODEL, lane);
        bf16x16 b1 = load_frag_b16(wtk + k0, D_MODEL, lane);
        bf16x16 b2 = load_frag_b16(wtv + k0, D_MODEL, lane);
        accq = wmma_bf16(a, b0, accq);
        acck = wmma_bf16(a, b1, acck);
        accv = wmma_bf16(a, b2, accv);
    }

    const int h    = (lane >> 4) & 1;
    const int col  = lane & 15;
    const float biasq = bq[n0 + col];
    const float biask = bk[n0 + col];
    const float biasv = bv[n0 + col];
    const float qscale = 0.125f;  // 1/sqrt(d_head)

#pragma unroll
    for (int r = 0; r < 8; ++r) {
        int row = row0 + r + 8 * h;
        Qb[(size_t)row * D_HEAD + n0 + col] = (__bf16)((accq[r] + biasq) * qscale);
        Kb[(size_t)row * D_HEAD + n0 + col] = (__bf16)(acck[r] + biask);
        int bb = row >> 12;              // row / SEQ
        int s  = row & (SEQ - 1);        // row % SEQ
        Vt[((size_t)bb * D_HEAD + n0 + col) * SEQ + s] = (__bf16)(accv[r] + biasv);
    }
}

// ---------------- Kernel 3: causal flash attention (WMMA bf16) ------------
// grid.x = NROWS/64; 4 waves per block, each owns a 16-query tile.
// K/V tiles (64 keys) are staged cooperatively with async global->LDS copies,
// double-buffered so block kb+1 streams in while kb computes.
__global__ void __launch_bounds__(128)
flash_attn_kernel(const __bf16* __restrict__ Qb, const __bf16* __restrict__ Kb,
                  const __bf16* __restrict__ Vt, float* __restrict__ out) {
    __shared__ __align__(16) __bf16 ktile[2][KBLK][D_HEAD];  // 2 x 8KB: [key][dh]
    __shared__ __align__(16) __bf16 vtile[2][D_HEAD][KBLK];  // 2 x 8KB: [dh][key]
    __shared__ __align__(16) __bf16 lds_p[4][16][KBLK];      // 8KB P relayout

    const int tid   = threadIdx.x;
    const int lane  = tid & 31;
    const int wave  = tid >> 5;
    const int rowblk = blockIdx.x * 64;
    const int row0  = rowblk + wave * 16;          // global query row
    const int bb    = row0 >> 12;
    const int q0    = row0 & (SEQ - 1);            // this wave's tile seq pos
    const int q0base = rowblk & (SEQ - 1);         // block's base seq pos
    const int h     = (lane >> 4) & 1;
    const int cidx  = lane & 15;

    const __bf16* Kbb = Kb + (size_t)bb * SEQ * D_HEAD;
    const __bf16* Vtb = Vt + (size_t)bb * D_HEAD * SEQ;

    bf16x16 qa0 = load_frag_b16(Qb + (size_t)row0 * D_HEAD + 0,  D_HEAD, lane);
    bf16x16 qa1 = load_frag_b16(Qb + (size_t)row0 * D_HEAD + 32, D_HEAD, lane);

    f32x8 oacc[4] = {};
    float m[8], l[8];
#pragma unroll
    for (int r = 0; r < 8; ++r) { m[r] = -3.0e38f; l[r] = 0.0f; }

    // Uniform trip count across the 4 waves (derived from block base row):
    // last block has k0 = q0base <= q0 of every wave, so no fully-masked block.
    const int nkb = (q0base >> 6) + 1;

    // Stage block 0. Each thread copies 4 x 16B per tile.
    {
        const __bf16* ksrc = Kbb;                  // 64x64 contiguous
#pragma unroll
        for (int j = 0; j < 4; ++j) {
            int c = tid + 128 * j;                 // chunk 0..511
            async_copy16(ksrc + c * 8, &ktile[0][0][0] + c * 8);
        }
#pragma unroll
        for (int j = 0; j < 4; ++j) {
            int c = tid + 128 * j;
            int d = c >> 3, o = (c & 7) * 8;
            async_copy16(Vtb + (size_t)d * SEQ + o, &vtile[0][0][0] + c * 8);
        }
    }

    for (int kb = 0; kb < nkb; ++kb) {
        const int k0  = kb * KBLK;
        const int buf = kb & 1;

        asm volatile("s_wait_asynccnt 0x0" ::: "memory");  // my staging done
        __syncthreads();                                    // everyone's done

        if (kb + 1 < nkb) {                                 // prefetch next block
            const __bf16* ksrc = Kbb + (size_t)(k0 + KBLK) * D_HEAD;
#pragma unroll
            for (int j = 0; j < 4; ++j) {
                int c = tid + 128 * j;
                async_copy16(ksrc + c * 8, &ktile[buf ^ 1][0][0] + c * 8);
            }
#pragma unroll
            for (int j = 0; j < 4; ++j) {
                int c = tid + 128 * j;
                int d = c >> 3, o = (c & 7) * 8;
                async_copy16(Vtb + (size_t)d * SEQ + (k0 + KBLK) + o,
                             &vtile[buf ^ 1][0][0] + c * 8);
            }
        }

        // ---- scores: 4 key sub-tiles of 16, dh split 0..31 / 32..63 ----
        f32x8 s[4];
#pragma unroll
        for (int t = 0; t < 4; ++t) {
            bf16x16 b0 = load_frag_b16(&ktile[buf][t * 16][0],  D_HEAD, lane);
            bf16x16 b1 = load_frag_b16(&ktile[buf][t * 16][32], D_HEAD, lane);
            f32x8 st = {};
            st = wmma_bf16(qa0, b0, st);
            st = wmma_bf16(qa1, b1, st);
            s[t] = st;
        }

        if (k0 + (KBLK - 1) > q0) {      // diagonal block: causal mask
#pragma unroll
            for (int t = 0; t < 4; ++t)
#pragma unroll
                for (int r = 0; r < 8; ++r) {
                    int qi = q0 + r + 8 * h;
                    if (k0 + 16 * t + cidx > qi) s[t][r] = -3.0e38f;
                }
        }

        // ---- online softmax ----
        float alpha[8];
#pragma unroll
        for (int r = 0; r < 8; ++r) {
            float mx = fmaxf(fmaxf(s[0][r], s[1][r]), fmaxf(s[2][r], s[3][r]));
            float mb = fmaxf(rowmax16(mx), m[r]);
            float p0 = __expf(s[0][r] - mb);
            float p1 = __expf(s[1][r] - mb);
            float p2 = __expf(s[2][r] - mb);
            float p3 = __expf(s[3][r] - mb);
            float a  = __expf(m[r] - mb);
            m[r] = mb;
            l[r] = l[r] * a + rowsum16(p0 + p1 + p2 + p3);
            alpha[r] = a;
            int rr = r + 8 * h;
            lds_p[wave][rr][cidx]      = (__bf16)p0;
            lds_p[wave][rr][cidx + 16] = (__bf16)p1;
            lds_p[wave][rr][cidx + 32] = (__bf16)p2;
            lds_p[wave][rr][cidx + 48] = (__bf16)p3;
        }
#pragma unroll
        for (int r = 0; r < 8; ++r) {
            oacc[0][r] *= alpha[r]; oacc[1][r] *= alpha[r];
            oacc[2][r] *= alpha[r]; oacc[3][r] *= alpha[r];
        }

        // LDS ops are wave-ordered: drain P stores, reload as A fragments.
        asm volatile("s_wait_dscnt 0x0" ::: "memory");
        bf16x16 pa0 = load_frag_b16(&lds_p[wave][0][0],  KBLK, lane);
        bf16x16 pa1 = load_frag_b16(&lds_p[wave][0][32], KBLK, lane);

        // ---- PV: V^T tile rows are dh columns -> contiguous key runs ----
#pragma unroll
        for (int nt = 0; nt < 4; ++nt) {
            bf16x16 v0 = load_frag_b16(&vtile[buf][nt * 16][0],  KBLK, lane);
            bf16x16 v1 = load_frag_b16(&vtile[buf][nt * 16][32], KBLK, lane);
            oacc[nt] = wmma_bf16(pa0, v0, oacc[nt]);
            oacc[nt] = wmma_bf16(pa1, v1, oacc[nt]);
        }
    }

#pragma unroll
    for (int r = 0; r < 8; ++r) {
        float inv = 1.0f / l[r];
        size_t orow = (size_t)(row0 + r + 8 * h) * D_HEAD;
        out[orow +  0 + cidx] = oacc[0][r] * inv;
        out[orow + 16 + cidx] = oacc[1][r] * inv;
        out[orow + 32 + cidx] = oacc[2][r] * inv;
        out[orow + 48 + cidx] = oacc[3][r] * inv;
    }
}

extern "C" void kernel_launch(void* const* d_in, const int* in_sizes, int n_in,
                              void* d_out, int out_size, void* d_ws, size_t ws_size,
                              hipStream_t stream) {
    const float* x  = (const float*)d_in[0];
    const float* Wq = (const float*)d_in[1];
    const float* bq = (const float*)d_in[2];
    const float* Wk = (const float*)d_in[3];
    const float* bk = (const float*)d_in[4];
    const float* Wv = (const float*)d_in[5];
    const float* bv = (const float*)d_in[6];
    float* out = (float*)d_out;

    char* ws = (char*)d_ws;
    const size_t wt_bytes  = (size_t)3 * D_HEAD * D_MODEL * 2;   // 393216
    const size_t qkv_bytes = (size_t)NROWS * D_HEAD * 2;         // 2097152
    __bf16* Wt = (__bf16*)ws;
    __bf16* Qb = (__bf16*)(ws + wt_bytes);
    __bf16* Kb = (__bf16*)(ws + wt_bytes + qkv_bytes);
    __bf16* Vt = (__bf16*)(ws + wt_bytes + 2 * qkv_bytes);

    wt_prep_kernel<<<(3 * D_HEAD * D_MODEL + 255) / 256, 256, 0, stream>>>(Wq, Wk, Wv, Wt);
    qkv_proj_kernel<<<NROWS / 16, 128, 0, stream>>>(x, Wt, bq, bk, bv, Qb, Kb, Vt);
    flash_attn_kernel<<<NROWS / 64, 128, 0, stream>>>(Qb, Kb, Vt, out);
}
